// MultiHeadAttention_2035814498424
// MI455X (gfx1250) — matrix operand order repaired and checked by reading, not compiled, not run on hardware
//
#include <hip/hip_runtime.h>
#include <hip/hip_bf16.h>

// ---------------------------------------------------------------------------
// MI455X (gfx1250) fused multi-head attention, bf16 WMMA pipeline.
//   x,W (f32) --convert--> bf16
//   Q = x@Wq^T+bq (bf16), K likewise, V stored transposed per head,
//   flash-attention per 16-row query tile (v_wmma_f32_16x16x32_bf16),
//   out = O@Wp^T+bp (f32).
// GEMM: constant shapes (8192x768x768) -> sub-tile strides fold into 24-bit
// immediate offsets; 32x64 tile/wave = 8 WMMA per K-step (~21 FLOP/byte).
// __launch_bounds__(.,1) gives the allocator the full VGPR file: no spills.
// ---------------------------------------------------------------------------

typedef __attribute__((ext_vector_type(16))) __bf16 v16bf;
typedef __attribute__((ext_vector_type(8)))  float  v8f;

#define BATCH   8
#define SEQ     1024
#define EMB     768
#define HEADS   8
#define DHEAD   96
#define HEADBLK (SEQ * DHEAD)   // 98304 elements per head block
#define MROWS   (BATCH * SEQ)   // 8192

__device__ __forceinline__ unsigned short f2bf(float f) {
  unsigned int u = __float_as_uint(f);
  u += 0x7FFFu + ((u >> 16) & 1u);          // round-to-nearest-even
  return (unsigned short)(u >> 16);
}

union FragAB { v16bf v; uint4 u[2]; };      // 32 bytes = one WMMA A/B fragment

// ---------------------------------------------------------------------------
// Packed convert: 8 f32 -> 8 bf16 per thread (b128 in, b128 out).
__global__ void convert_f32_bf16x8(const float4* __restrict__ src,
                                   uint4* __restrict__ dst, int n8) {
  int i = blockIdx.x * blockDim.x + threadIdx.x;
  int stride = gridDim.x * blockDim.x;
  for (; i < n8; i += stride) {
    float4 f0 = src[2 * i], f1 = src[2 * i + 1];
    uint4 o;
    o.x = (unsigned)f2bf(f0.x) | ((unsigned)f2bf(f0.y) << 16);
    o.y = (unsigned)f2bf(f0.z) | ((unsigned)f2bf(f0.w) << 16);
    o.z = (unsigned)f2bf(f1.x) | ((unsigned)f2bf(f1.y) << 16);
    o.w = (unsigned)f2bf(f1.z) | ((unsigned)f2bf(f1.w) << 16);
    dst[i] = o;
  }
}

// ---------------------------------------------------------------------------
// Y[8192 x 768] = A[8192 x 768] @ W[768 x 768]^T + bias.  One wave -> 32x64.
// MODE 0: f32 out (final projection)
// MODE 1: bf16 out, natural [row][col] layout (Q, K)
// MODE 2: bf16 out, per-head transposed V: Vt[b][h][96][1024]
template <int MODE>
__global__ void __launch_bounds__(256, 1)
gemm_bf16_wmma(const unsigned short* __restrict__ A,
               const unsigned short* __restrict__ W,
               const float* __restrict__ bias,
               void* __restrict__ out) {
  constexpr int KD = EMB;                        // 768, compile-time
  constexpr int NC = EMB;
  constexpr int tilesN = NC / 64;                // 12
  const int lane = threadIdx.x & 31;
  const int wave = threadIdx.x >> 5;
  const int tile = blockIdx.x * 8 + wave;        // grid exact: no tail guard
  const int tm = tile / tilesN, tn = tile % tilesN;
  const int r  = lane & 15;                      // A row / B col within tile
  const int hf = lane >> 4;                      // lane half-group

  // Two base pointers; every other operand address is an immediate offset.
  const unsigned short* ap = A + (size_t)(tm * 32 + r) * KD + hf * 16;
  const unsigned short* bp = W + (size_t)(tn * 64 + r) * KD + hf * 16;

  v8f acc[2][4];
#pragma unroll
  for (int i = 0; i < 2; ++i)
#pragma unroll
    for (int j = 0; j < 4; ++j) acc[i][j] = (v8f){};

#pragma unroll 1
  for (int k0 = 0; k0 < KD; k0 += 32) {
    FragAB a[2], b[4];
    a[0].u[0] = *(const uint4*)(ap + k0);
    a[0].u[1] = *(const uint4*)(ap + k0 + 8);
    a[1].u[0] = *(const uint4*)(ap + k0 + 16 * KD);
    a[1].u[1] = *(const uint4*)(ap + k0 + 16 * KD + 8);
#pragma unroll
    for (int j = 0; j < 4; ++j) {
      b[j].u[0] = *(const uint4*)(bp + k0 + j * 16 * KD);
      b[j].u[1] = *(const uint4*)(bp + k0 + j * 16 * KD + 8);
    }
    if (k0 + 64 < KD) {                          // uniform -> EXEC unchanged
      __builtin_prefetch(ap + k0 + 64, 0, 1);
      __builtin_prefetch(bp + k0 + 64, 0, 1);
    }
#pragma unroll
    for (int i = 0; i < 2; ++i)
#pragma unroll
      for (int j = 0; j < 4; ++j)
        acc[i][j] = __builtin_amdgcn_wmma_f32_16x16x32_bf16(
            false, a[i].v, false, b[j].v, (short)0, acc[i][j], false, false);
  }

#pragma unroll
  for (int j = 0; j < 4; ++j) {
    const int col = tn * 64 + j * 16 + r;
    const float bv = bias[col];
#pragma unroll
    for (int i = 0; i < 2; ++i) {
#pragma unroll
      for (int v = 0; v < 8; ++v) {
        const int row = tm * 32 + i * 16 + v + 8 * hf;  // C layout: VGPR v
        const float val = acc[i][j][v] + bv;
        if constexpr (MODE == 0) {
          ((float*)out)[(size_t)row * NC + col] = val;
        } else if constexpr (MODE == 1) {
          ((unsigned short*)out)[(size_t)row * NC + col] = f2bf(val);
        } else {                                        // Vt[b][h][96][1024]
          int bb  = row >> 10, n = row & 1023;
          int f   = n * EMB + col;                      // within-batch flat
          int hh  = f / HEADBLK;
          int rem = f - hh * HEADBLK;
          int np  = rem / DHEAD;
          int dd  = rem - np * DHEAD;
          ((unsigned short*)out)[(((size_t)(bb * HEADS + hh) * DHEAD + dd) << 10)
                                 + np] = f2bf(val);
        }
      }
    }
  }
}

// ---------------------------------------------------------------------------
// Flash attention: 4 waves/block share one (b, h); each wave owns a 16-row
// query tile -> K/V streams hit L1/L2 across the block. 32-key blocks.
__global__ void __launch_bounds__(128, 1)
attention_wmma(const unsigned short* __restrict__ Q,
               const unsigned short* __restrict__ K,
               const unsigned short* __restrict__ Vt,
               unsigned short* __restrict__ O) {
  __shared__ unsigned short lds[4 * 16 * 32];  // per-wave 16x32 bf16 P staging
  const int lane = threadIdx.x & 31;
  const int wv   = threadIdx.x >> 5;
  const int bh   = blockIdx.x >> 4;            // (b,h) pair: 64 total
  const int qt   = (blockIdx.x & 15) * 4 + wv; // query tile 0..63
  const int h    = bh & 7;
  const int b    = bh >> 3;
  const int r    = lane & 15;
  const int hf   = lane >> 4;

  const unsigned short* qh = Q  + (size_t)b * (SEQ * EMB) + h * HEADBLK;
  const unsigned short* kh = K  + (size_t)b * (SEQ * EMB) + h * HEADBLK;
  const unsigned short* vh = Vt + (size_t)(b * HEADS + h) * DHEAD * SEQ;
  unsigned short*       oh = O  + (size_t)b * (SEQ * EMB) + h * HEADBLK;

  // Q block 16x96 as three 16x32 A-fragments, resident for the whole pass.
  FragAB qf[3];
#pragma unroll
  for (int t = 0; t < 3; ++t) {
    const unsigned short* p = qh + (size_t)(qt * 16 + r) * DHEAD + t * 32 + hf * 16;
    qf[t].u[0] = *(const uint4*)p;
    qf[t].u[1] = *(const uint4*)(p + 8);
  }

  v8f oacc[6];
#pragma unroll
  for (int i = 0; i < 6; ++i) oacc[i] = (v8f){};
  float m[8], l[8];
#pragma unroll
  for (int v = 0; v < 8; ++v) { m[v] = -3.0e38f; l[v] = 0.0f; }

  unsigned short* pl = lds + wv * 512;

#pragma unroll 1
  for (int kb = 0; kb < SEQ; kb += 32) {
    // ---- S = Q @ K^T for 32 keys (two 16-col tiles, 3 K-steps over d=96)
    v8f s0 = {}, s1 = {};
#pragma unroll
    for (int t = 0; t < 3; ++t) {
      FragAB b0, b1;
      const unsigned short* p0 = kh + (size_t)(kb + r) * DHEAD + t * 32 + hf * 16;
      const unsigned short* p1 = p0 + 16 * DHEAD;
      b0.u[0] = *(const uint4*)p0;  b0.u[1] = *(const uint4*)(p0 + 8);
      b1.u[0] = *(const uint4*)p1;  b1.u[1] = *(const uint4*)(p1 + 8);
      s0 = __builtin_amdgcn_wmma_f32_16x16x32_bf16(false, qf[t].v, false, b0.v,
                                                   (short)0, s0, false, false);
      s1 = __builtin_amdgcn_wmma_f32_16x16x32_bf16(false, qf[t].v, false, b1.v,
                                                   (short)0, s1, false, false);
    }

    // ---- online softmax; row v+8*hf lives across the 16 lanes of a half.
    // exp values are stored straight to LDS (transpose staging) -> no arrays.
#pragma unroll
    for (int v = 0; v < 8; ++v) {
      float mb = fmaxf(s0[v], s1[v]);
      mb = fmaxf(mb, __shfl_xor(mb, 1));
      mb = fmaxf(mb, __shfl_xor(mb, 2));
      mb = fmaxf(mb, __shfl_xor(mb, 4));
      mb = fmaxf(mb, __shfl_xor(mb, 8));
      float mn = fmaxf(m[v], mb);
      float sc = __expf(m[v] - mn);
      m[v] = mn;
      float e0 = __expf(s0[v] - mn);
      float e1 = __expf(s1[v] - mn);
      int row = v + 8 * hf;
      pl[row * 32 + r]      = f2bf(e0);
      pl[row * 32 + 16 + r] = f2bf(e1);
      float rs = e0 + e1;
      rs += __shfl_xor(rs, 1);
      rs += __shfl_xor(rs, 2);
      rs += __shfl_xor(rs, 4);
      rs += __shfl_xor(rs, 8);
      l[v] = l[v] * sc + rs;
#pragma unroll
      for (int nt = 0; nt < 6; ++nt) oacc[nt][v] = oacc[nt][v] * sc;
    }

    // ---- transpose P (C layout -> A layout) via wave-private LDS
    asm volatile("s_wait_dscnt 0" ::: "memory");   // same-wave LDS RAW fence
    FragAB pf;
    pf.u[0] = *(const uint4*)(pl + r * 32 + hf * 16);
    pf.u[1] = *(const uint4*)(pl + r * 32 + hf * 16 + 8);

    // ---- O += P @ V_block  (Vt rows are d-cols -> contiguous B fragments)
#pragma unroll
    for (int nt = 0; nt < 6; ++nt) {
      FragAB vb;
      const unsigned short* vp = vh + (size_t)(nt * 16 + r) * SEQ + kb + hf * 16;
      vb.u[0] = *(const uint4*)vp;
      vb.u[1] = *(const uint4*)(vp + 8);
      oacc[nt] = __builtin_amdgcn_wmma_f32_16x16x32_bf16(
          false, pf.v, false, vb.v, (short)0, oacc[nt], false, false);
    }
  }

  // softmax divided by sqrt(EMB) AFTER softmax (faithful) -> fold into 1/l.
  const float scale = 0.03608439182435161f;   // 1/sqrt(768)
#pragma unroll
  for (int v = 0; v < 8; ++v) {
    float inv = scale / l[v];
    int row = qt * 16 + v + 8 * hf;
#pragma unroll
    for (int nt = 0; nt < 6; ++nt) {
      oh[(size_t)row * DHEAD + nt * 16 + r] = f2bf(oacc[nt][v] * inv);
    }
  }
}

// ---------------------------------------------------------------------------
extern "C" void kernel_launch(void* const* d_in, const int* in_sizes, int n_in,
                              void* d_out, int out_size, void* d_ws, size_t ws_size,
                              hipStream_t stream) {
  const float* x  = (const float*)d_in[0];
  const float* Wq = (const float*)d_in[1];
  const float* bq = (const float*)d_in[2];
  const float* Wk = (const float*)d_in[3];
  const float* bk = (const float*)d_in[4];
  const float* Wv = (const float*)d_in[5];
  const float* bv = (const float*)d_in[6];
  const float* Wp = (const float*)d_in[7];
  const float* bp = (const float*)d_in[8];

  const size_t NX = (size_t)BATCH * SEQ * EMB;  // 6,291,456
  const size_t NW = (size_t)EMB * EMB;          //   589,824

  unsigned short* ws  = (unsigned short*)d_ws;
  unsigned short* xb  = ws;  ws += NX;
  unsigned short* wqb = ws;  ws += NW;
  unsigned short* wkb = ws;  ws += NW;
  unsigned short* wvb = ws;  ws += NW;
  unsigned short* wpb = ws;  ws += NW;
  unsigned short* Qb  = ws;  ws += NX;
  unsigned short* Kb  = ws;  ws += NX;
  unsigned short* Vt  = ws;  ws += NX;
  unsigned short* Ob  = ws;  ws += NX;

  convert_f32_bf16x8<<<1024, 256, 0, stream>>>((const float4*)x,  (uint4*)xb,
                                               (int)(NX / 8));
  convert_f32_bf16x8<<<288, 256, 0, stream>>>((const float4*)Wq, (uint4*)wqb,
                                              (int)(NW / 8));
  convert_f32_bf16x8<<<288, 256, 0, stream>>>((const float4*)Wk, (uint4*)wkb,
                                              (int)(NW / 8));
  convert_f32_bf16x8<<<288, 256, 0, stream>>>((const float4*)Wv, (uint4*)wvb,
                                              (int)(NW / 8));
  convert_f32_bf16x8<<<288, 256, 0, stream>>>((const float4*)Wp, (uint4*)wpb,
                                              (int)(NW / 8));

  const int gblocks = (MROWS / 32) * (EMB / 64) / 8;  // 3072 tiles / 8 waves

  gemm_bf16_wmma<1><<<gblocks, 256, 0, stream>>>(xb, wqb, bq, Qb);
  gemm_bf16_wmma<1><<<gblocks, 256, 0, stream>>>(xb, wkb, bk, Kb);
  gemm_bf16_wmma<2><<<gblocks, 256, 0, stream>>>(xb, wvb, bv, Vt);

  attention_wmma<<<1024, 128, 0, stream>>>(Qb, Kb, Vt, Ob);

  gemm_bf16_wmma<0><<<gblocks, 256, 0, stream>>>(Ob, wpb, bp, d_out);
}
